// RNNs_29703993819861
// MI455X (gfx1250) — matrix operand order repaired and checked by reading, not compiled, not run on hardware
//
#include <hip/hip_runtime.h>

typedef __attribute__((ext_vector_type(16))) _Float16 v16h;
typedef __attribute__((ext_vector_type(8)))  _Float16 v8h;
typedef __attribute__((ext_vector_type(8)))  float    v8f;
typedef __attribute__((ext_vector_type(2)))  float    v2f;

#define Bsz   16384
#define Tlen  256
#define INs   3
#define Hs    32
#define OUTs  3
#define PREDL 300

// CDNA5 hardware tanh (trans op). Fallback: raw v_tanh_f32 + v_nop (TRANS RAW hazard).
__device__ __forceinline__ float tanh_hw(float x) {
#if __has_builtin(__builtin_amdgcn_tanhf)
    return __builtin_amdgcn_tanhf(x);
#else
    float y;
    asm("v_tanh_f32 %0, %1\n\tv_nop" : "=v"(y) : "v"(x));
    return y;
#endif
}
__device__ __forceinline__ float sig_hw(float x) {
    // sigmoid(x) = 0.5*tanh(x/2) + 0.5
    return __builtin_fmaf(tanh_hw(x * 0.5f), 0.5f, 0.5f);
}

extern "C" __global__ void __launch_bounds__(256)
lstm_s2s(const float* __restrict__ xin,    // [B, T, 3]
         const float* __restrict__ W_ih,   // [128, 3]
         const float* __restrict__ W_hh,   // [128, 32]
         const float* __restrict__ b_ih,   // [128]
         const float* __restrict__ b_hh,   // [128]
         const float* __restrict__ W_out,  // [3, 32]
         const float* __restrict__ b_out,  // [3]
         float* __restrict__ out)          // [B, 300, 3]
{
    // Per-wave LDS slice (1536 B): h tile [16 rows][40 halves] (80B stride,
    // conflict-free ds_load_b128), then y tile [16 rows][4 floats] at byte 1280.
    __shared__ __align__(16) _Float16 smem[8 * 768];

    const int tid     = threadIdx.x;
    const int lane    = tid & 31;
    const int wave    = tid >> 5;
    const int row16   = lane & 15;      // N/col index in B/C/D layouts; M/row in A
    const int hi      = lane >> 4;      // 0: lanes 0-15, 1: lanes 16-31
    const int k0      = hi << 4;        // K base for f16 B-operand halves
    const int rowBase = (blockIdx.x * 8 + wave) << 4;

    _Float16* sm = smem + wave * 768;
    float*    yb = (float*)(sm + 640);  // y feedback buffer [16][4] f32

    { // zero this wave's LDS slice once
        unsigned* w = (unsigned*)sm;
        #pragma unroll
        for (int i = lane; i < 384; i += 32) w[i] = 0u;
    }

    // ---- W_hh -> f16 WMMA B layout: v16h element e <-> K = k0+e, N = 16t+row16 ----
    v16h Bhh[8];
    #pragma unroll
    for (int t = 0; t < 8; ++t) {
        const int n = 16 * t + row16;
        const float* wr = W_hh + n * Hs + k0;     // 16 contiguous floats of row n
        #pragma unroll
        for (int e = 0; e < 16; ++e) Bhh[t][e] = (_Float16)wr[e];
    }

    // ---- W_ih + combined bias -> f32 K=4 WMMA B layout ----
    // A K-rows are [x0, x1, x2, 1.0]; B: VGPR0 = K(0|2), VGPR1 = K(1|3) by lane half.
    v2f Bxb[8];
    #pragma unroll
    for (int t = 0; t < 8; ++t) {
        const int n = 16 * t + row16;
        const float w0 = W_ih[n * 3 + 0];
        const float w1 = W_ih[n * 3 + 1];
        const float w2 = W_ih[n * 3 + 2];
        const float bg = b_ih[n] + b_hh[n];
        Bxb[t][0] = hi ? w2 : w0;   // K=2 : K=0
        Bxb[t][1] = hi ? bg : w1;   // K=3 : K=1
    }

    // ---- W_out -> f16 B layout (cols 3..15 zero) ----
    v16h Bout;
    #pragma unroll
    for (int e = 0; e < 16; ++e) Bout[e] = (_Float16)0.f;
    if (row16 < OUTs) {
        const float* wr = W_out + row16 * Hs + k0;
        #pragma unroll
        for (int e = 0; e < 16; ++e) Bout[e] = (_Float16)wr[e];
    }
    const float bo = (row16 < OUTs) ? b_out[row16] : 0.f;

    // ---- recurrent state ----
    v8f c0v, c1v;       // cell state, D layout (hidden 0-15 / 16-31)
    v16h Ah;            // h as f16 A operand (16x32)
    v8f zero8;
    #pragma unroll
    for (int v = 0; v < 8; ++v) { c0v[v] = 0.f; c1v[v] = 0.f; zero8[v] = 0.f; }
    #pragma unroll
    for (int e = 0; e < 16; ++e) Ah[e] = (_Float16)0.f;

    // One LSTM step. Axb = f32 A operand [x0,x1,x2,1]:
    //   gates = Axb*Bxb (f32 K=4: x@Wih.T + bias)  +  Ah*Bhh (f16 K=32)
    auto cell = [&](v2f Axb) {
        v8f acc[8];
        #pragma unroll
        for (int t = 0; t < 8; ++t)
            acc[t] = __builtin_amdgcn_wmma_f32_16x16x4_f32(
                         false, Axb, false, Bxb[t], (short)0, zero8, false, false);
        #pragma unroll
        for (int t = 0; t < 8; ++t)
            acc[t] = __builtin_amdgcn_wmma_f32_16x16x32_f16(
                         false, Ah, false, Bhh[t], (short)0, acc[t], false, false);
        // gate order i,f,g,o -> tiles {0,1},{2,3},{4,5},{6,7}; p = hidden half
        #pragma unroll
        for (int p = 0; p < 2; ++p) {
            v8f& cc = p ? c1v : c0v;
            _Float16* hp = sm + (8 * hi) * 40 + 16 * p + row16;
            #pragma unroll
            for (int v = 0; v < 8; ++v) {
                const float iv = sig_hw(acc[0 + p][v]);
                const float fv = sig_hw(acc[2 + p][v]);
                const float gv = tanh_hw(acc[4 + p][v]);
                const float ov = sig_hw(acc[6 + p][v]);
                const float cs = __builtin_fmaf(fv, cc[v], iv * gv);
                cc[v] = cs;
                hp[v * 40] = (_Float16)(ov * tanh_hw(cs));  // row v+8hi, col 16p+row16
            }
        }
        // rebuild A-layout h: lane row=row16, K = {16hi..16hi+15}
        const v8h* base = (const v8h*)(sm + row16 * 40 + hi * 16);
        const v8h lo = base[0];
        const v8h hh = base[1];
        #pragma unroll
        for (int e = 0; e < 8; ++e) { Ah[e] = lo[e]; Ah[8 + e] = hh[e]; }
    };

    // y = h @ W_out.T + b_out; store to global; return next f32 A operand [y0,y1,y2,1]
    auto emit_y = [&](int p) -> v2f {
        v8f accy;
        #pragma unroll
        for (int v = 0; v < 8; ++v) accy[v] = bo;
        accy = __builtin_amdgcn_wmma_f32_16x16x32_f16(
                   false, Ah, false, Bout, (short)0, accy, false, false);
        if (row16 < OUTs) {
            float* op = out + (size_t)(rowBase + 8 * hi) * (PREDL * OUTs)
                            + (size_t)p * OUTs + row16;
            float* yp = yb + (8 * hi) * 4 + row16;
            #pragma unroll
            for (int v = 0; v < 8; ++v) {
                __builtin_nontemporal_store(accy[v], op + v * (PREDL * OUTs));
                yp[v * 4] = accy[v];            // row v+8hi
            }
        }
        v2f ax;
        if (!hi) { ax[0] = yb[row16 * 4 + 0]; ax[1] = yb[row16 * 4 + 1]; }
        else     { ax[0] = yb[row16 * 4 + 2]; ax[1] = 1.0f; }
        return ax;
    };

    // ---- encoder: 256 steps, x load software-pipelined one step ahead ----
    float xa0, xa1;
    {
        const float* xr = xin + (size_t)(rowBase + row16) * (Tlen * INs);
        if (!hi) { xa0 = xr[0]; xa1 = xr[1]; }
        else     { xa0 = xr[2]; xa1 = 1.0f; }
    }
    for (int t = 0; t < Tlen; ++t) {
        float na0 = xa0, na1 = xa1;
        if (t + 1 < Tlen) {
            const float* xr = xin + ((size_t)(rowBase + row16) * Tlen + (t + 1)) * INs;
            if (!hi) { na0 = xr[0]; na1 = xr[1]; }
            else     { na0 = xr[2]; na1 = 1.0f; }
        }
        v2f ax; ax[0] = xa0; ax[1] = xa1;
        cell(ax);
        xa0 = na0; xa1 = na1;
    }

    // ---- decoder: y0 from final encoder h, then 299 autoregressive steps ----
    v2f ay = emit_y(0);
    for (int p = 1; p < PREDL; ++p) {
        cell(ay);
        ay = emit_y(p);
    }
}

extern "C" void kernel_launch(void* const* d_in, const int* in_sizes, int n_in,
                              void* d_out, int out_size, void* d_ws, size_t ws_size,
                              hipStream_t stream) {
    const float* xin   = (const float*)d_in[0];
    const float* W_ih  = (const float*)d_in[1];
    const float* W_hh  = (const float*)d_in[2];
    const float* b_ih  = (const float*)d_in[3];
    const float* b_hh  = (const float*)d_in[4];
    const float* W_out = (const float*)d_in[5];
    const float* b_out = (const float*)d_in[6];
    float* out = (float*)d_out;
    (void)in_sizes; (void)n_in; (void)out_size; (void)d_ws; (void)ws_size;

    // 16 batch rows per wave, 8 waves per block -> 128 rows/block, 128 blocks.
    dim3 grid(Bsz / 128);
    dim3 block(256);
    hipLaunchKernelGGL(lstm_s2s, grid, block, 0, stream,
                       xin, W_ih, W_hh, b_ih, b_hh, W_out, b_out, out);
}
